// MEGNet_74818330296973
// MI455X (gfx1250) — matrix operand order repaired and checked by reading, not compiled, not run on hardware
//
#include <hip/hip_runtime.h>
#include <hip/hip_bf16.h>

// ---------------------------------------------------------------------------
// CDNA5 (gfx1250) wave32 WMMA types
// ---------------------------------------------------------------------------
typedef __attribute__((ext_vector_type(16))) __bf16 v16bf;
typedef __attribute__((ext_vector_type(8)))  float  v8f;

union FragBF { unsigned int u[8]; uint4 q[2]; v16bf v; };

__device__ __forceinline__ unsigned short f2bf(float f) {
    unsigned int x = __float_as_uint(f);
    // round-to-nearest-even truncation to bf16
    unsigned int r = x + 0x7FFFu + ((x >> 16) & 1u);
    return (unsigned short)(r >> 16);
}
__device__ __forceinline__ unsigned int pack_bf2(float lo, float hi) {
    return (unsigned int)f2bf(lo) | ((unsigned int)f2bf(hi) << 16);
}

__device__ __forceinline__ void copy16_f4(const float* __restrict__ p, float* d) {
    const float4* q = (const float4*)p;   // p is 16B-aligned by construction
#pragma unroll
    for (int j = 0; j < 4; ++j) {
        float4 v = q[j];
        d[4 * j + 0] = v.x; d[4 * j + 1] = v.y;
        d[4 * j + 2] = v.z; d[4 * j + 3] = v.w;
    }
}

// ---------------------------------------------------------------------------
// Row loaders: fetch 16 consecutive K elements for one row.
// OOB rows are CLAMPED (not zeroed): WMMA output row m depends only on A row
// m, and the epilogue masks stores for rows >= M, so clamped garbage is safe.
// Only the K-tail needs zero-fill, and that test is wave-uniform.
// k0 is always a multiple of 16, so concat loaders never cross a 64-segment.
// ---------------------------------------------------------------------------
struct DirectLd {
    const float* X; int K; int M;
    __device__ void fetch16(int row, int k0, float* d) const {
        int r = row < M ? row : M - 1;          // branchless clamp
        const float* p = X + (size_t)r * (size_t)K;
        if (k0 + 16 <= K) {                     // uniform branch
            if ((K & 3) == 0) {
                copy16_f4(p + k0, d);           // 16B-aligned
            } else if ((K & 1) == 0) {
                const float2* q = (const float2*)(p + k0);
#pragma unroll
                for (int j = 0; j < 8; ++j) {
                    float2 v = q[j];
                    d[2 * j + 0] = v.x; d[2 * j + 1] = v.y;
                }
            } else {
#pragma unroll
                for (int i = 0; i < 16; ++i) d[i] = p[k0 + i];
            }
        } else {                                // uniform tail: zero-fill
#pragma unroll
            for (int i = 0; i < 16; ++i) d[i] = (k0 + i < K) ? p[k0 + i] : 0.0f;
        }
    }
};

// comb_e row = [x_t[src] | x_t[dst] | e_t | u_t[batch[src]]], K = 256
struct EdgeCatLd {
    const float* Xt; const float* Et; const float* Ut;
    const int* src; const int* dst; const int* batch; int M;
    __device__ void fetch16(int row, int k0, float* d) const {
        int r = row < M ? row : M - 1;          // branchless clamp
        int seg = k0 >> 6, kk = k0 & 63;
        const float* p;
        if (seg == 0)      p = Xt + (size_t)src[r] * 64 + kk;
        else if (seg == 1) p = Xt + (size_t)dst[r] * 64 + kk;
        else if (seg == 2) p = Et + (size_t)r * 64 + kk;
        else               p = Ut + (size_t)batch[src[r]] * 64 + kk;
        copy16_f4(p, d);
    }
};

// row = [A[row] | B[row] | C[idxC ? idxC[row] : row]], K = 192
struct Cat3Ld {
    const float* A; const float* B; const float* C; const int* idxC; int M;
    __device__ void fetch16(int row, int k0, float* d) const {
        int r = row < M ? row : M - 1;          // branchless clamp
        int seg = k0 >> 6, kk = k0 & 63;
        const float* p;
        if (seg == 0)      p = A + (size_t)r * 64 + kk;
        else if (seg == 1) p = B + (size_t)r * 64 + kk;
        else               p = C + (size_t)(idxC ? idxC[r] : r) * 64 + kk;
        copy16_f4(p, d);
    }
};

// ---------------------------------------------------------------------------
// WMMA GEMM: Y[M,64] = relu(Row(m) . W[K,64] + bias[64])
// Block = 256 threads = 8 wave32; tile 128 rows x 64 cols; K stepped by 32.
// Each wave computes one 16-row strip = 4x v_wmma_f32_16x16x32_bf16 per step.
// Bias is pre-loaded into the C accumulator; ReLU fused in epilogue.
// LDS rows have stride 20 uints (80B, 16B-aligned) so all LDS traffic is b128.
// ---------------------------------------------------------------------------
#define LDS_STRIDE 20

template <class LD>
__global__ __launch_bounds__(256) void gemm_wmma(
    LD ld, const float* __restrict__ W, const float* __restrict__ bias,
    float* __restrict__ Y, int M, int K)
{
    __shared__ alignas(16) unsigned int Xs[128 * LDS_STRIDE];  // 128 rows x 32 K (bf16 pairs)
    __shared__ alignas(16) unsigned int Wts[64 * LDS_STRIDE];  // W^T: 64 cols x 32 K

    const int tid  = threadIdx.x;
    const int wave = tid >> 5;
    const int lane = tid & 31;
    const int half = lane >> 4;
    const int l15  = lane & 15;
    const int rowBase = blockIdx.x * 128;
    const int r0 = rowBase + wave * 16;

    // Accumulators initialized with broadcast bias (C operand of WMMA).
    v8f acc[4];
#pragma unroll
    for (int t = 0; t < 4; ++t) {
        float bv = bias[t * 16 + l15];
#pragma unroll
        for (int j = 0; j < 8; ++j) acc[t][j] = bv;
    }

    const int nsteps = (K + 31) >> 5;
    for (int s = 0; s < nsteps; ++s) {
        const int k0 = s << 5;
        __syncthreads();
        // ---- stage A-tile (rows of X, gathered/concatenated by loader) ----
        {
            int row = tid >> 1;          // 0..127
            int kh  = tid & 1;           // which 16-wide half of the 32-K step
            float tmp[16];
            ld.fetch16(rowBase + row, k0 + kh * 16, tmp);
            unsigned int pk[8];
#pragma unroll
            for (int j = 0; j < 8; ++j) pk[j] = pack_bf2(tmp[2 * j], tmp[2 * j + 1]);
            uint4* dst = (uint4*)&Xs[row * LDS_STRIDE + kh * 8];
            dst[0] = make_uint4(pk[0], pk[1], pk[2], pk[3]);
            dst[1] = make_uint4(pk[4], pk[5], pk[6], pk[7]);
        }
        // ---- stage B-tile transposed: Wts[n][k] ----
        {
            int n  = tid & 63;
            int pb = (tid >> 6) * 4;     // 4 k-pairs per thread
            unsigned int pk[4];
#pragma unroll
            for (int q = 0; q < 4; ++q) {
                int k = k0 + 2 * (pb + q);
                float lo = (k     < K) ? W[(size_t)k * 64 + n]       : 0.0f;
                float hi = (k + 1 < K) ? W[(size_t)(k + 1) * 64 + n] : 0.0f;
                pk[q] = pack_bf2(lo, hi);
            }
            *(uint4*)&Wts[n * LDS_STRIDE + pb] = make_uint4(pk[0], pk[1], pk[2], pk[3]);
        }
        __syncthreads();

        // ---- A fragment: 16x32 bf16 (ISA 7.12.2 layout): lanes 0-15 hold
        //      K[0:8)+K[16:24), lanes 16-31 hold K[8:16)+K[24:32) ----
        FragBF a;
        {
            const unsigned int* xr = &Xs[(wave * 16 + l15) * LDS_STRIDE];
            a.q[0] = *(const uint4*)(xr + half * 4);
            a.q[1] = *(const uint4*)(xr + 8 + half * 4);
        }
        // ---- B fragments for all 4 N-tiles, then back-to-back WMMAs ----
        FragBF b[4];
#pragma unroll
        for (int t = 0; t < 4; ++t) {
            const unsigned int* wr = &Wts[(t * 16 + l15) * LDS_STRIDE];
            b[t].q[0] = *(const uint4*)(wr + half * 4);
            b[t].q[1] = *(const uint4*)(wr + 8 + half * 4);
        }
#pragma unroll
        for (int t = 0; t < 4; ++t) {
            acc[t] = __builtin_amdgcn_wmma_f32_16x16x32_bf16(
                false, a.v, false, b[t].v, (short)0, acc[t], false, false);
        }
    }

    // ---- epilogue: ReLU + store (C/D layout: M = j + 8*half, N = lane&15) --
#pragma unroll
    for (int t = 0; t < 4; ++t) {
#pragma unroll
        for (int j = 0; j < 8; ++j) {
            int row = r0 + j + 8 * half;
            if (row < M) {
                float v = acc[t][j];
                v = v > 0.0f ? v : 0.0f;
                Y[(size_t)row * 64 + t * 16 + l15] = v;
            }
        }
    }
}

// ---------------------------------------------------------------------------
// BatchNorm1d (training mode, biased variance, affine), 64 columns.
// ---------------------------------------------------------------------------
__global__ void bn_stats_k(const float* __restrict__ X, float* __restrict__ st, int M)
{
    __shared__ float sb[256], qb[256];
    int col = threadIdx.x & 63;
    int rg  = threadIdx.x >> 6;            // 0..3
    int base = blockIdx.x * 1024;
    int end = base + 1024; if (end > M) end = M;
    float s = 0.0f, ss = 0.0f;
    for (int r = base + rg; r < end; r += 4) {
        float v = X[(size_t)r * 64 + col];
        s += v; ss += v * v;
    }
    sb[threadIdx.x] = s; qb[threadIdx.x] = ss;
    __syncthreads();
    if (rg == 0) {
        s  = sb[col] + sb[64 + col] + sb[128 + col] + sb[192 + col];
        ss = qb[col] + qb[64 + col] + qb[128 + col] + qb[192 + col];
        atomicAdd(&st[col], s);
        atomicAdd(&st[64 + col], ss);
    }
}

__global__ void bn_apply_k(float* __restrict__ X, const float* __restrict__ st,
                           const float* __restrict__ g, const float* __restrict__ beta,
                           int M)
{
    int t = blockIdx.x * blockDim.x + threadIdx.x;
    if (t >= M * 64) return;
    int c = t & 63;
    float invM = 1.0f / (float)M;
    float m = st[c] * invM;
    float v = st[64 + c] * invM - m * m;
    float inv = rsqrtf(v + 1e-5f);
    X[t] = (X[t] - m) * inv * g[c] + beta[c];
}

// ---------------------------------------------------------------------------
// segment mean (scatter_mean, empty segments -> 0 via count clamp to 1)
// ---------------------------------------------------------------------------
__global__ void seg_scatter_k(const float* __restrict__ X, const int* __restrict__ idx,
                              float* __restrict__ acc, int M)
{
    int t = blockIdx.x * blockDim.x + threadIdx.x;
    if (t >= M * 64) return;
    int r = t >> 6, c = t & 63;
    atomicAdd(&acc[(size_t)idx[r] * 64 + c], X[t]);
}
__global__ void seg_count_k(const int* __restrict__ idx, float* __restrict__ cnt, int M)
{
    int t = blockIdx.x * blockDim.x + threadIdx.x;
    if (t < M) atomicAdd(&cnt[idx[t]], 1.0f);
}
__global__ void seg_div_k(const float* __restrict__ acc, const float* __restrict__ cnt,
                          float* __restrict__ out, int S)
{
    int t = blockIdx.x * blockDim.x + threadIdx.x;
    if (t >= S * 64) return;
    int r = t >> 6;
    float c = cnt[r];
    out[t] = acc[t] / (c > 1.0f ? c : 1.0f);
}

// ---------------------------------------------------------------------------
// elementwise residual add: y = a + b (y may alias b)
// ---------------------------------------------------------------------------
__global__ void add_k(const float* __restrict__ a, const float* __restrict__ b,
                      float* __restrict__ y, int n)
{
    int t = blockIdx.x * blockDim.x + threadIdx.x;
    if (t < n) y[t] = a[t] + b[t];
}

// final linear [G,64] @ [64,1] + b
__global__ void out_k(const float* __restrict__ X, const float* __restrict__ W,
                      const float* __restrict__ b, float* __restrict__ y, int G)
{
    int t = blockIdx.x * blockDim.x + threadIdx.x;
    if (t >= G) return;
    float s = b[0];
#pragma unroll
    for (int k = 0; k < 64; ++k) s += X[(size_t)t * 64 + k] * W[k];
    y[t] = s;
}

// ---------------------------------------------------------------------------
// Host orchestration
// ---------------------------------------------------------------------------
extern "C" void kernel_launch(void* const* d_in, const int* in_sizes, int n_in,
                              void* d_out, int out_size, void* d_ws, size_t ws_size,
                              hipStream_t stream)
{
    // ---- parameter index map (JAX pytree leaf order: dict keys sorted) ----
    // params keys sorted: layers, out, post, pre
    // layer keys sorted:  e_embed, edge, glob, node, u_embed, x_embed
    // mlp keys sorted:    bn (g,beta x3), lin (W,b x3)
    struct LayerIdx { int ee[4], ed_bn[6], ed_lin[6], gl_bn[6], gl_lin[6],
                          nd_bn[6], nd_lin[6], ue[4], xe[4]; };
    LayerIdx LI[3];
    int P = 0;
    for (int i = 0; i < 3; ++i) {
        for (int j = 0; j < 4; ++j) LI[i].ee[j]     = P++;
        for (int j = 0; j < 6; ++j) LI[i].ed_bn[j]  = P++;
        for (int j = 0; j < 6; ++j) LI[i].ed_lin[j] = P++;
        for (int j = 0; j < 6; ++j) LI[i].gl_bn[j]  = P++;
        for (int j = 0; j < 6; ++j) LI[i].gl_lin[j] = P++;
        for (int j = 0; j < 6; ++j) LI[i].nd_bn[j]  = P++;
        for (int j = 0; j < 6; ++j) LI[i].nd_lin[j] = P++;
        for (int j = 0; j < 4; ++j) LI[i].ue[j]     = P++;
        for (int j = 0; j < 4; ++j) LI[i].xe[j]     = P++;
    }
    const int iOutW = P++, iOutB = P++;
    const int iPostW = P++, iPostB = P++;
    const int iPreW = P++, iPreB = P++;
    const int iX = P++, iEA = P++, iU = P++, iEI = P++, iB = P++;
    (void)n_in; (void)out_size;

    auto F = [&](int i) { return (const float*)d_in[i]; };
    const int Nn = in_sizes[iX]  / 100;   // F_NODE = 100
    const int E  = in_sizes[iEA] / 50;    // F_EDGE = 50
    const int G  = in_sizes[iU]  / 3;     // F_U    = 3
    const float* x_in  = F(iX);
    const float* ea_in = F(iEA);
    const float* u_in  = F(iU);
    const int* src   = (const int*)d_in[iEI];
    const int* dst   = src + E;
    const int* batch = (const int*)d_in[iB];

    // ---- workspace layout ----
    float* w = (float*)d_ws; size_t off = 0; (void)ws_size;
    auto alloc = [&](size_t n) { float* p = w + off; off += n; return p; };
    const size_t nf = (size_t)Nn * 64, ef = (size_t)E * 64, gf = (size_t)G * 64;
    float* xh   = alloc(nf); float* x_t  = alloc(nf); float* xa = alloc(nf);
    float* xb   = alloc(nf); float* v_e  = alloc(nf);
    float* nacc = alloc(nf); float* ncnt = alloc((size_t)Nn);
    float* eh   = alloc(ef); float* e_t  = alloc(ef);
    float* ea   = alloc(ef); float* eb   = alloc(ef);
    float* uh   = alloc(gf); float* u_t  = alloc(gf); float* ua = alloc(gf);
    float* ub   = alloc(gf); float* u_e  = alloc(gf); float* u_v = alloc(gf);
    float* xpool= alloc(gf); float* epool= alloc(gf);
    float* gacc = alloc(gf); float* gcnt = alloc((size_t)G);
    float* stats= alloc(128); float* post1 = alloc(gf);

    // ---- launch helpers ----
    auto gemmD = [&](const float* X, int M, int K, const float* W, const float* b,
                     float* Y) {
        DirectLd ld{X, K, M};
        gemm_wmma<DirectLd><<<dim3((M + 127) / 128), 256, 0, stream>>>(ld, W, b, Y, M, K);
    };
    auto gemmE = [&](const EdgeCatLd& ld, int M, const float* W, const float* b,
                     float* Y) {
        gemm_wmma<EdgeCatLd><<<dim3((M + 127) / 128), 256, 0, stream>>>(ld, W, b, Y, M, 256);
    };
    auto gemmC = [&](const Cat3Ld& ld, int M, const float* W, const float* b,
                     float* Y) {
        gemm_wmma<Cat3Ld><<<dim3((M + 127) / 128), 256, 0, stream>>>(ld, W, b, Y, M, 192);
    };
    auto bn = [&](float* X, int M, const float* g, const float* beta) {
        hipMemsetAsync(stats, 0, 128 * sizeof(float), stream);
        bn_stats_k<<<dim3((M + 1023) / 1024), 256, 0, stream>>>(X, stats, M);
        bn_apply_k<<<dim3((M * 64 + 255) / 256), 256, 0, stream>>>(X, stats, g, beta, M);
    };
    auto seg_mean = [&](const float* X, const int* idx, int M, int S,
                        float* acc, float* cnt, float* out) {
        hipMemsetAsync(acc, 0, (size_t)S * 64 * sizeof(float), stream);
        hipMemsetAsync(cnt, 0, (size_t)S * sizeof(float), stream);
        seg_scatter_k<<<dim3((M * 64 + 255) / 256), 256, 0, stream>>>(X, idx, acc, M);
        seg_count_k<<<dim3((M + 255) / 256), 256, 0, stream>>>(idx, cnt, M);
        seg_div_k<<<dim3((S * 64 + 255) / 256), 256, 0, stream>>>(acc, cnt, out, S);
    };
    auto addv = [&](const float* a, const float* b, float* y, size_t n) {
        add_k<<<dim3(((int)n + 255) / 256), 256, 0, stream>>>(a, b, y, (int)n);
    };

    // ---- forward ----
    gemmD(x_in, Nn, 100, F(iPreW), F(iPreB), xh);              // pre: [N,100]->[N,64] + ReLU

    for (int i = 0; i < 3; ++i) {
        const LayerIdx& L = LI[i];
        // embeds (Linear+ReLU x2)
        gemmD(xh, Nn, 64, F(L.xe[0]), F(L.xe[1]), xa);
        gemmD(xa, Nn, 64, F(L.xe[2]), F(L.xe[3]), x_t);
        const float* ein = (i == 0) ? ea_in : eh; int eK = (i == 0) ? 50 : 64;
        gemmD(ein, E, eK, F(L.ee[0]), F(L.ee[1]), ea);
        gemmD(ea, E, 64, F(L.ee[2]), F(L.ee[3]), e_t);
        const float* uin = (i == 0) ? u_in : uh; int uK = (i == 0) ? 3 : 64;
        gemmD(uin, G, uK, F(L.ue[0]), F(L.ue[1]), ua);
        gemmD(ua, G, 64, F(L.ue[2]), F(L.ue[3]), u_t);
        // edge MLP on fused comb_e = [x_t[src]|x_t[dst]|e_t|u_t[batch[src]]]
        EdgeCatLd eld{x_t, e_t, u_t, src, dst, batch, E};
        gemmE(eld, E, F(L.ed_lin[0]), F(L.ed_lin[1]), ea); bn(ea, E, F(L.ed_bn[0]), F(L.ed_bn[1]));
        gemmD(ea, E, 64, F(L.ed_lin[2]), F(L.ed_lin[3]), eb); bn(eb, E, F(L.ed_bn[2]), F(L.ed_bn[3]));
        gemmD(eb, E, 64, F(L.ed_lin[4]), F(L.ed_lin[5]), ea); bn(ea, E, F(L.ed_bn[4]), F(L.ed_bn[5]));
        // node update
        seg_mean(ea, src, E, Nn, nacc, ncnt, v_e);
        Cat3Ld nld{x_t, v_e, u_t, batch, Nn};
        gemmC(nld, Nn, F(L.nd_lin[0]), F(L.nd_lin[1]), xa); bn(xa, Nn, F(L.nd_bn[0]), F(L.nd_bn[1]));
        gemmD(xa, Nn, 64, F(L.nd_lin[2]), F(L.nd_lin[3]), xb); bn(xb, Nn, F(L.nd_bn[2]), F(L.nd_bn[3]));
        gemmD(xb, Nn, 64, F(L.nd_lin[4]), F(L.nd_lin[5]), xa); bn(xa, Nn, F(L.nd_bn[4]), F(L.nd_bn[5]));
        // global update
        seg_mean(v_e, batch, Nn, G, gacc, gcnt, u_e);
        seg_mean(xa, batch, Nn, G, gacc, gcnt, u_v);
        Cat3Ld gld{u_e, u_v, u_t, nullptr, G};
        gemmC(gld, G, F(L.gl_lin[0]), F(L.gl_lin[1]), ua); bn(ua, G, F(L.gl_bn[0]), F(L.gl_bn[1]));
        gemmD(ua, G, 64, F(L.gl_lin[2]), F(L.gl_lin[3]), ub); bn(ub, G, F(L.gl_bn[2]), F(L.gl_bn[3]));
        gemmD(ub, G, 64, F(L.gl_lin[4]), F(L.gl_lin[5]), ua); bn(ua, G, F(L.gl_bn[4]), F(L.gl_bn[5]));
        // residuals
        if (i == 0) {
            addv(xa, x_t, xh, nf); addv(ea, e_t, eh, ef); addv(ua, u_t, uh, gf);
        } else {
            addv(xa, xh, xh, nf); addv(ea, eh, eh, ef); addv(ua, uh, uh, gf);
        }
    }

    // pooling + readout
    seg_mean(xh, batch, Nn, G, gacc, gcnt, xpool);
    seg_mean(eh, src, E, Nn, nacc, ncnt, xa);           // node-level mean of eh
    seg_mean(xa, batch, Nn, G, gacc, gcnt, epool);
    Cat3Ld pld{xpool, epool, uh, nullptr, G};
    gemmC(pld, G, F(iPostW), F(iPostB), post1);          // [G,192]->[G,64] + ReLU
    out_k<<<dim3((G + 127) / 128), 128, 0, stream>>>(post1, F(iOutW), F(iOutB),
                                                     (float*)d_out, G);
}
